// BidirLSTMModel_60911226192291
// MI455X (gfx1250) — matrix operand order repaired and checked by reading, not compiled, not run on hardware
//
#include <hip/hip_runtime.h>
#include <math.h>

// Problem constants
#define B_   256
#define T_   128
#define E_   128
#define D_   512
#define V_   50000
#define G4D  2048      // 4*D
#define D2   1024      // 2*D
#define VPAD 50048     // V rounded up to multiple of 128 (BN)

// GEMM tiling
#define BM 128
#define BN 128
#define BK 32

typedef __attribute__((ext_vector_type(16))) _Float16 v16h;
typedef __attribute__((ext_vector_type(8)))  float    v8f;

__device__ __forceinline__ float sigmf(float x) { return 1.0f / (1.0f + __expf(-x)); }

// ---------------- async global->LDS staging (CDNA5 ASYNCcnt path) ----------------
#if defined(__AMDGCN__) && __has_builtin(__builtin_amdgcn_global_load_async_to_lds_b128)
#define ASYNC_LDS 1
#else
#define ASYNC_LDS 0
#endif

#if ASYNC_LDS
typedef int vi4 __attribute__((ext_vector_type(4)));
typedef __attribute__((address_space(3))) vi4 vi4_lds;

// Flat LDS addresses carry the LDS byte offset in the low 32 bits (ISA aperture
// rule: the upper address bits are dropped).
__device__ __forceinline__ void async_ld16(void* lds, const void* g) {
  __builtin_amdgcn_global_load_async_to_lds_b128(
      (vi4*)g,
      (vi4_lds*)(unsigned)(unsigned long long)lds,
      0, 0);
}
__device__ __forceinline__ void async_wait0() {
#if __has_builtin(__builtin_amdgcn_s_wait_asynccnt)
  __builtin_amdgcn_s_wait_asynccnt(0);
#else
  asm volatile("s_wait_asynccnt 0x0" ::: "memory");
#endif
}
#endif

// Stage 32B (16 halfs) per thread for A and B tiles into LDS.
__device__ __forceinline__ void stage_tiles(_Float16* as, _Float16* bs,
                                            const _Float16* ga, const _Float16* gb,
                                            int lrow, int lcol) {
#if ASYNC_LDS
  async_ld16(&as[lrow * BK + lcol],     ga);
  async_ld16(&as[lrow * BK + lcol + 8], ga + 8);
  async_ld16(&bs[lrow * BK + lcol],     gb);
  async_ld16(&bs[lrow * BK + lcol + 8], gb + 8);
#else
  const uint4* pa = (const uint4*)ga;
  const uint4* pb = (const uint4*)gb;
  uint4 a0 = pa[0], a1 = pa[1], b0 = pb[0], b1 = pb[1];
  *(uint4*)&as[lrow * BK + lcol]     = a0;
  *(uint4*)&as[lrow * BK + lcol + 8] = a1;
  *(uint4*)&bs[lrow * BK + lcol]     = b0;
  *(uint4*)&bs[lrow * BK + lcol + 8] = b1;
#endif
}

__device__ __forceinline__ void stage_fence() {
#if ASYNC_LDS
  async_wait0();
#endif
  __syncthreads();
}

// ---------------- elementwise / prep kernels ----------------

__global__ void k_embed(const int* __restrict__ tokens, const float* __restrict__ emb,
                        _Float16* __restrict__ x, int nrows) {
  int i = blockIdx.x * blockDim.x + threadIdx.x;
  if (i >= nrows * E_) return;
  int row = i / E_, e = i - row * E_;
  x[i] = (_Float16)emb[(size_t)tokens[row] * E_ + e];
}

// in f32 [R,C] row-major  ->  out f16 [C,R] row-major (out row stride = R)
__global__ void k_transpose(const float* __restrict__ in, _Float16* __restrict__ out,
                            int R, int C) {
  long long i = (long long)blockIdx.x * blockDim.x + threadIdx.x;
  if (i >= (long long)R * C) return;
  int r = (int)(i / C), c = (int)(i - (long long)r * C);
  out[(size_t)c * R + r] = (_Float16)in[i];
}

// Gate-interleaved transpose: in f32 [K, 4D] (cols = g*D + d), out f16 [4D, K]
// with output row n_new = 4*d + g.
__global__ void k_transpose_gate(const float* __restrict__ in, _Float16* __restrict__ out,
                                 int K) {
  long long i = (long long)blockIdx.x * blockDim.x + threadIdx.x;
  if (i >= (long long)K * G4D) return;
  int k = (int)(i / G4D), n_old = (int)(i - (long long)k * G4D);
  int g = n_old / D_, d = n_old - g * D_;
  int n_new = 4 * d + g;
  out[(size_t)n_new * K + k] = (_Float16)in[i];
}

// bias permute: out[4*d+g] = in[g*D+d]
__global__ void k_permute_bias(const float* __restrict__ in, float* __restrict__ out) {
  int i = blockIdx.x * blockDim.x + threadIdx.x;
  if (i >= G4D) return;
  int g = i / D_, d = i - g * D_;
  out[4 * d + g] = in[i];
}

__global__ void k_cvt(const float* __restrict__ in, _Float16* __restrict__ out, int n) {
  int i = blockIdx.x * blockDim.x + threadIdx.x;
  if (i < n) out[i] = (_Float16)in[i];
}

__global__ void k_pack_state(const float* __restrict__ c, _Float16* __restrict__ st, int col0) {
  int i = blockIdx.x * blockDim.x + threadIdx.x;
  if (i >= B_ * D_) return;
  int b = i / D_, d = i - b * D_;
  st[(size_t)b * D2 + col0 + d] = (_Float16)c[i];
}

// ---------------- WMMA fragment helpers ----------------
__device__ __forceinline__ v16h frag_a(const _Float16* __restrict__ p, int ld, int lane) {
  int row = lane & 15;
  int ko  = (lane >> 4) << 3;
  const _Float16* q = p + row * ld + ko;
  v16h r;
#pragma unroll
  for (int i = 0; i < 8; ++i) { r[i] = q[i]; r[i + 8] = q[i + 16]; }
  return r;
}

__device__ __forceinline__ v16h frag_b(const _Float16* __restrict__ p, int ld, int lane) {
  int n  = lane & 15;
  int kb = (lane >> 4) << 3;
  const _Float16* q = p + n * ld + kb;
  v16h r;
#pragma unroll
  for (int i = 0; i < 8; ++i) { r[i] = q[i]; r[i + 8] = q[i + 16]; }
  return r;
}

// ---------------- generic WMMA GEMM ----------------
// C[M,N] f32 = A[M,K] f16 (row-major) x BT[N,K] f16 (row-major) + bias[N]; optional relu.
__global__ __launch_bounds__(256)
void k_gemm(const _Float16* __restrict__ A, const _Float16* __restrict__ Bt,
            const float* __restrict__ bias, float* __restrict__ C,
            int M, int N, int K, int relu) {
  __shared__ __align__(16) _Float16 As[2][BM * BK];
  __shared__ __align__(16) _Float16 Bs[2][BN * BK];
  const int bm   = blockIdx.y * BM;
  const int bn   = blockIdx.x * BN;
  const int tid  = threadIdx.x;
  const int lane = tid & 31;
  const int wid  = tid >> 5;
  const int wm   = (wid & 3) * 32;
  const int wn   = (wid >> 2) * 64;
  const int lrow = tid >> 1;
  const int lcol = (tid & 1) * 16;

  v8f acc[2][4] = {};

  const _Float16* gA = A  + (size_t)(bm + lrow) * K + lcol;
  const _Float16* gB = Bt + (size_t)(bn + lrow) * K + lcol;
  stage_tiles(As[0], Bs[0], gA, gB, lrow, lcol);
  stage_fence();

  for (int k0 = 0; k0 < K; k0 += BK) {
    const int buf = (k0 / BK) & 1;
    if (k0 + BK < K) {
      const int nb = buf ^ 1;
      stage_tiles(As[nb], Bs[nb], gA + k0 + BK, gB + k0 + BK, lrow, lcol);
      if (k0 + 2 * BK < K) {  // L2 prefetch of the K+2 tile
        __builtin_prefetch(gA + k0 + 2 * BK, 0, 1);
        __builtin_prefetch(gB + k0 + 2 * BK, 0, 1);
      }
    }
    v16h af[2], bf[4];
#pragma unroll
    for (int fm = 0; fm < 2; ++fm)
      af[fm] = frag_a(&As[buf][(wm + fm * 16) * BK], BK, lane);
#pragma unroll
    for (int fn = 0; fn < 4; ++fn)
      bf[fn] = frag_b(&Bs[buf][(wn + fn * 16) * BK], BK, lane);
#pragma unroll
    for (int fm = 0; fm < 2; ++fm)
#pragma unroll
      for (int fn = 0; fn < 4; ++fn)
        acc[fm][fn] = __builtin_amdgcn_wmma_f32_16x16x32_f16(
            false, af[fm], false, bf[fn], (short)0, acc[fm][fn], false, false);
    stage_fence();
  }

  const int frow = lane & 15;
  const int mh   = (lane >> 4) * 8;
#pragma unroll
  for (int fm = 0; fm < 2; ++fm) {
#pragma unroll
    for (int fn = 0; fn < 4; ++fn) {
      int gn = bn + wn + fn * 16 + frow;
      if (gn < N) {
        float bv = bias ? bias[gn] : 0.0f;
        int gm = bm + wm + fm * 16 + mh;
#pragma unroll
        for (int r = 0; r < 8; ++r) {
          float v = acc[fm][fn][r] + bv;
          if (relu) v = fmaxf(v, 0.0f);
          C[(size_t)(gm + r) * N + gn] = v;
        }
      }
    }
  }
}

// ---------------- fused recurrent GEMM + LSTM cell step ----------------
// Gate-interleaved layout: WhT/xg columns are n = 4*d + g, so the 128-wide N tile
// covers 32 hidden units x 4 gates -> cell update is block-local.
__global__ __launch_bounds__(256)
void k_lstm_fused(const _Float16* __restrict__ Hin, const _Float16* __restrict__ WhT,
                  const float* __restrict__ xg, const int* __restrict__ lengths,
                  int t, int rev, _Float16* __restrict__ Hout, float* __restrict__ c,
                  _Float16* __restrict__ out, int ocol) {
  __shared__ __align__(16) _Float16 As[2][BM * BK];
  __shared__ __align__(16) _Float16 Bs[2][BN * BK];
  __shared__ __align__(16) _Float16 Gs[BM * BN];
  const int bm   = blockIdx.y * BM;
  const int bn   = blockIdx.x * BN;
  const int tid  = threadIdx.x;
  const int lane = tid & 31;
  const int wid  = tid >> 5;
  const int wm   = (wid & 3) * 32;
  const int wn   = (wid >> 2) * 64;
  const int lrow = tid >> 1;
  const int lcol = (tid & 1) * 16;
  const int K = D_;

  v8f acc[2][4] = {};

  const _Float16* gA = Hin + (size_t)(bm + lrow) * K + lcol;
  const _Float16* gB = WhT + (size_t)(bn + lrow) * K + lcol;
  stage_tiles(As[0], Bs[0], gA, gB, lrow, lcol);
  stage_fence();

  for (int k0 = 0; k0 < K; k0 += BK) {
    const int buf = (k0 / BK) & 1;
    if (k0 + BK < K) {
      const int nb = buf ^ 1;
      stage_tiles(As[nb], Bs[nb], gA + k0 + BK, gB + k0 + BK, lrow, lcol);
      if (k0 + 2 * BK < K) {
        __builtin_prefetch(gA + k0 + 2 * BK, 0, 1);
        __builtin_prefetch(gB + k0 + 2 * BK, 0, 1);
      }
    }
    v16h af[2], bf[4];
#pragma unroll
    for (int fm = 0; fm < 2; ++fm)
      af[fm] = frag_a(&As[buf][(wm + fm * 16) * BK], BK, lane);
#pragma unroll
    for (int fn = 0; fn < 4; ++fn)
      bf[fn] = frag_b(&Bs[buf][(wn + fn * 16) * BK], BK, lane);
#pragma unroll
    for (int fm = 0; fm < 2; ++fm)
#pragma unroll
      for (int fn = 0; fn < 4; ++fn)
        acc[fm][fn] = __builtin_amdgcn_wmma_f32_16x16x32_f16(
            false, af[fm], false, bf[fn], (short)0, acc[fm][fn], false, false);
    stage_fence();
  }

  // Deposit gate pre-activations (+ gathered xg row) into the LDS gate tile.
  const int frow = lane & 15;
  const int mh   = (lane >> 4) * 8;
#pragma unroll
  for (int fm = 0; fm < 2; ++fm) {
#pragma unroll
    for (int fn = 0; fn < 4; ++fn) {
      int cl = wn + fn * 16 + frow;
      int gn = bn + cl;
#pragma unroll
      for (int r = 0; r < 8; ++r) {
        int rl = wm + fm * 16 + mh + r;
        int b  = bm + rl;
        int len = lengths[b];
        int st  = rev ? (t < len ? len - 1 - t : t) : t;
        Gs[rl * BN + cl] =
            (_Float16)(acc[fm][fn][r] + xg[((size_t)b * T_ + st) * G4D + gn]);
      }
    }
  }
  __syncthreads();

  // Cell update: rows [bm,bm+128) x hidden units [bn/4, bn/4+32).
  const int d0 = bn >> 2;
  for (int idx = tid; idx < BM * 32; idx += 256) {
    int rl = idx >> 5, dl = idx & 31;
    int b = bm + rl, d = d0 + dl;
    float gi = (float)Gs[rl * BN + 4 * dl + 0];
    float gj = (float)Gs[rl * BN + 4 * dl + 1];
    float gf = (float)Gs[rl * BN + 4 * dl + 2];
    float go = (float)Gs[rl * BN + 4 * dl + 3];
    size_t ci = (size_t)b * D_ + d;
    float cc = c[ci];
    float cn = sigmf(gf + 1.0f) * cc + sigmf(gi) * tanhf(gj);
    float hn = sigmf(go) * tanhf(cn);
    int  len = lengths[b];
    bool m   = t < len;
    c[ci]    = m ? cn : cc;
    Hout[ci] = m ? (_Float16)hn : Hin[ci];
    if (out) {
      int st = rev ? (m ? len - 1 - t : t) : t;
      out[((size_t)b * T_ + st) * D2 + ocol + d] = m ? (_Float16)hn : (_Float16)0.0f;
    }
  }
}

// ---------------- host orchestration ----------------
extern "C" void kernel_launch(void* const* d_in, const int* in_sizes, int n_in,
                              void* d_out, int out_size, void* d_ws, size_t ws_size,
                              hipStream_t stream) {
  const int*   tokens  = (const int*)d_in[0];
  const int*   lengths = (const int*)d_in[1];
  const float* emb     = (const float*)d_in[2];
  const float* Wx[4]   = {(const float*)d_in[3], (const float*)d_in[6],
                          (const float*)d_in[9], (const float*)d_in[12]};
  const float* Wh[4]   = {(const float*)d_in[4], (const float*)d_in[7],
                          (const float*)d_in[10], (const float*)d_in[13]};
  const float* bg[4]   = {(const float*)d_in[5], (const float*)d_in[8],
                          (const float*)d_in[11], (const float*)d_in[14]};
  const float* W_head  = (const float*)d_in[15];
  const float* b1      = (const float*)d_in[16];
  const float* U       = (const float*)d_in[17];
  const float* b2      = (const float*)d_in[18];
  float* pred = (float*)d_out;

  char* ws = (char*)d_ws;
  size_t off = 0;
  auto alloc = [&](size_t bytes) -> char* {
    char* p = ws + off;
    off += (bytes + 255) & ~(size_t)255;
    return p;
  };

  _Float16* x0    = (_Float16*)alloc((size_t)B_ * T_ * E_ * 2);
  float*    xg    = (float*)   alloc((size_t)B_ * T_ * G4D * 4);
  _Float16* out0  = (_Float16*)alloc((size_t)B_ * T_ * D2 * 2);
  _Float16* WxT[4];
  WxT[0] = (_Float16*)alloc((size_t)G4D * E_ * 2);
  WxT[1] = (_Float16*)alloc((size_t)G4D * E_ * 2);
  WxT[2] = (_Float16*)alloc((size_t)G4D * D2 * 2);
  WxT[3] = (_Float16*)alloc((size_t)G4D * D2 * 2);
  _Float16* WhT[4];
  for (int i = 0; i < 4; ++i) WhT[i] = (_Float16*)alloc((size_t)G4D * D_ * 2);
  float* bperm[4];
  for (int i = 0; i < 4; ++i) bperm[i] = (float*)alloc((size_t)G4D * 4);
  _Float16* WheadT = (_Float16*)alloc((size_t)D_ * D2 * 2);
  _Float16* UT     = (_Float16*)alloc((size_t)VPAD * D_ * 2);
  float*    c      = (float*)   alloc((size_t)B_ * D_ * 4);
  _Float16* h16a   = (_Float16*)alloc((size_t)B_ * D_ * 2);
  _Float16* h16b   = (_Float16*)alloc((size_t)B_ * D_ * 2);
  _Float16* st16   = (_Float16*)alloc((size_t)B_ * D2 * 2);
  float*    hh     = (float*)   alloc((size_t)B_ * D_ * 4);
  _Float16* hh16   = (_Float16*)alloc((size_t)B_ * D_ * 2);

  const int TPB = 256;
  auto cdiv = [](long long a, long long b) { return (int)((a + b - 1) / b); };

  // Weight conversions
  k_transpose_gate<<<cdiv((long long)E_ * G4D, TPB), TPB, 0, stream>>>(Wx[0], WxT[0], E_);
  k_transpose_gate<<<cdiv((long long)E_ * G4D, TPB), TPB, 0, stream>>>(Wx[1], WxT[1], E_);
  k_transpose_gate<<<cdiv((long long)D2 * G4D, TPB), TPB, 0, stream>>>(Wx[2], WxT[2], D2);
  k_transpose_gate<<<cdiv((long long)D2 * G4D, TPB), TPB, 0, stream>>>(Wx[3], WxT[3], D2);
  for (int i = 0; i < 4; ++i) {
    k_transpose_gate<<<cdiv((long long)D_ * G4D, TPB), TPB, 0, stream>>>(Wh[i], WhT[i], D_);
    k_permute_bias<<<cdiv(G4D, TPB), TPB, 0, stream>>>(bg[i], bperm[i]);
  }
  k_transpose<<<cdiv((long long)D2 * D_, TPB), TPB, 0, stream>>>(W_head, WheadT, D2, D_);
  k_transpose<<<cdiv((long long)D_ * V_, TPB), TPB, 0, stream>>>(U, UT, D_, V_);

  // Embedding gather -> f16
  k_embed<<<cdiv((long long)B_ * T_ * E_, TPB), TPB, 0, stream>>>(tokens, emb, x0, B_ * T_);

  auto run_lstm = [&](const _Float16* Ain, int Kin, const _Float16* WxTd, const float* bd,
                      const _Float16* WhTd, int rev, _Float16* outp, int ocol) {
    dim3 g1(G4D / BN, (B_ * T_) / BM);
    k_gemm<<<g1, TPB, 0, stream>>>(Ain, WxTd, bd, xg, B_ * T_, G4D, Kin, 0);
    (void)hipMemsetAsync(h16a, 0, (size_t)B_ * D_ * 2, stream);
    (void)hipMemsetAsync(c, 0, (size_t)B_ * D_ * 4, stream);
    dim3 g2(G4D / BN, B_ / BM);
    for (int t = 0; t < T_; ++t) {
      _Float16* hin  = (t & 1) ? h16b : h16a;
      _Float16* hout = (t & 1) ? h16a : h16b;
      k_lstm_fused<<<g2, TPB, 0, stream>>>(hin, WhTd, xg, lengths, t, rev, hout, c,
                                           outp, ocol);
    }
  };

  // Layer 0 (input = x0, K = E)
  run_lstm(x0, E_, WxT[0], bperm[0], WhT[0], 0, out0, 0);
  run_lstm(x0, E_, WxT[1], bperm[1], WhT[1], 1, out0, D_);
  // Layer 1 (input = out0, K = 2D); only final cell states needed
  run_lstm(out0, D2, WxT[2], bperm[2], WhT[2], 0, nullptr, 0);
  k_pack_state<<<cdiv((long long)B_ * D_, TPB), TPB, 0, stream>>>(c, st16, 0);
  run_lstm(out0, D2, WxT[3], bperm[3], WhT[3], 1, nullptr, 0);
  k_pack_state<<<cdiv((long long)B_ * D_, TPB), TPB, 0, stream>>>(c, st16, D_);

  // Head: relu(states @ W_head + b1) @ U + b2
  dim3 gh1(D_ / BN, B_ / BM);
  k_gemm<<<gh1, TPB, 0, stream>>>(st16, WheadT, b1, hh, B_, D_, D2, 1);
  k_cvt<<<cdiv((long long)B_ * D_, TPB), TPB, 0, stream>>>(hh, hh16, B_ * D_);
  dim3 gh2((V_ + BN - 1) / BN, B_ / BM);
  k_gemm<<<gh2, TPB, 0, stream>>>(hh16, UT, b2, pred, B_, V_, D_, 0);
}
